// RiemannianRefinement_29764123361416
// MI455X (gfx1250) — matrix operand order repaired and checked by reading, not compiled, not run on hardware
//
#include <hip/hip_runtime.h>
#include <hip/hip_bf16.h>
#include <math.h>

// ---------------------------------------------------------------------------
// Problem constants
// ---------------------------------------------------------------------------
#define BB 64
#define QQ 1024
#define DD 256
#define TT 64
#define SS 3
#define ROWS (BB * QQ)           // 65536

typedef __attribute__((ext_vector_type(16))) _Float16 v16h;
typedef __attribute__((ext_vector_type(8)))  _Float16 v8h;
typedef __attribute__((ext_vector_type(8)))  float    v8f;

// ---------------------------------------------------------------------------
// Workspace layout (bytes, 256-aligned)
// ---------------------------------------------------------------------------
#define OFF_NN     0u            // int[65536]                      -> 262144
#define OFF_SS     262144u       // float[64*3]                     -> 1024
#define OFF_ROT    263168u       // float[65536]                    -> 262144
#define OFF_TRN    525312u       // float[65536]                    -> 262144
#define OFF_MAT    787456u       // float[65536]                    -> 262144
#define OFF_RW1H   1049600u      // half[256*128]                   -> 65536
#define OFF_TW1H   1115136u      // half[256*128]                   -> 65536
#define OFF_MN1H   1180672u      // half[512*256]                   -> 262144

// ---------------------------------------------------------------------------
// Kernel W: convert layer-1 weights f32 -> f16 (row-major fin x fout, which is
// already the WMMA B-fragment-friendly layout: lane = N column, lanes read contiguous N)
// ---------------------------------------------------------------------------
__global__ void cvt_weights_kernel(const float* __restrict__ rw1,
                                   const float* __restrict__ tw1,
                                   const float* __restrict__ mn1,
                                   _Float16* __restrict__ rw1h,
                                   _Float16* __restrict__ tw1h,
                                   _Float16* __restrict__ mn1h) {
  int i = blockIdx.x * 256 + threadIdx.x;
  if (i < DD * (DD / 2)) {                 // 256*128
    rw1h[i] = (_Float16)rw1[i];
    tw1h[i] = (_Float16)tw1[i];
  }
  if (i < 2 * DD * DD) {                   // 512*256
    mn1h[i] = (_Float16)mn1[i];
  }
}

// ---------------------------------------------------------------------------
// Kernel A: step_scales = sigmoid(relu(mean_q(qf) @ sp_w1 + b1) @ sp_w2 + b2)
//           * clip(global_strength, 0, 1)      -- one block per batch
// ---------------------------------------------------------------------------
__global__ __launch_bounds__(256)
void scales_kernel(const float* __restrict__ qf,
                   const float* __restrict__ w1, const float* __restrict__ b1,
                   const float* __restrict__ w2, const float* __restrict__ b2,
                   const float* __restrict__ gsp, float* __restrict__ ss) {
  __shared__ float mean[DD];
  __shared__ float h[DD / 2];
  const int b = blockIdx.x, tid = threadIdx.x;
  const float* base = qf + (size_t)b * QQ * DD;
  float s = 0.f;
  for (int q = 0; q < QQ; ++q) s += base[(size_t)q * DD + tid];
  mean[tid] = s * (1.0f / (float)QQ);
  __syncthreads();
  if (tid < DD / 2) {
    float a = b1[tid];
    for (int d = 0; d < DD; ++d) a += mean[d] * w1[d * (DD / 2) + tid];
    h[tid] = fmaxf(a, 0.f);
  }
  __syncthreads();
  if (tid < SS) {
    float a = b2[tid];
    for (int j = 0; j < DD / 2; ++j) a += h[j] * w2[j * SS + tid];
    float gs = fminf(fmaxf(gsp[0], 0.f), 1.f);
    ss[b * SS + tid] = gs / (1.f + __expf(-a));
  }
}

// ---------------------------------------------------------------------------
// Kernel B: nearest-neighbor argmin over T=64 box centers (first-min ties)
// ---------------------------------------------------------------------------
__global__ __launch_bounds__(256)
void nn_kernel(const float* __restrict__ pb, const float* __restrict__ tbx,
               int* __restrict__ nn) {
  int row = blockIdx.x * 256 + threadIdx.x;
  if (row >= ROWS) return;
  int b = row >> 10;
  float cx = pb[(size_t)row * 4 + 0];
  float cy = pb[(size_t)row * 4 + 1];
  const float* tb = tbx + (size_t)b * TT * 4;
  float best = 3.402823e38f;
  int bi = 0;
  for (int t = 0; t < TT; ++t) {
    float dx = cx - tb[t * 4 + 0];
    float dy = cy - tb[t * 4 + 1];
    float d = dx * dx + dy * dy;
    if (d < best) { best = d; bi = t; }
  }
  nn[row] = bi;
}

// ---------------------------------------------------------------------------
// Per-wave fused MLP: 16 rows, hidden = NT*16, K = KS*32.
// Layer-1 via v_wmma_f32_16x16x32_f16; layer-2 (hidden -> 1) folded in-register
// and reduced across the 16 N-lanes with shfl_xor, then sigmoid.
// A fragment (ISA 7.12.2): lane m=lane&15, half=lane>>4;
//   f16 elem e: K = (e<8 ? half*8+e : 16+half*8+(e-8))  -> two 16B LDS chunks.
// B fragment: lane holds column N=lane&15 of 32x16 tile, 16 K values in A's K order.
// C/D: elem r -> (M = r+8*half, N = lane&15).
// ---------------------------------------------------------------------------
template <int NT, int KS>
__device__ __forceinline__ void mlp_wave(const _Float16* xs,  // LDS [16][256]
                                         const _Float16* ms,  // LDS [16][256] (KS==16)
                                         const _Float16* __restrict__ w1h,
                                         const float* __restrict__ b1,
                                         const float* __restrict__ w2,
                                         float b2, float* __restrict__ out,
                                         int row0, int lane) {
  const int m = lane & 15;
  const int half = lane >> 4;
  constexpr int N = NT * 16;
  float dot[8] = {0.f, 0.f, 0.f, 0.f, 0.f, 0.f, 0.f, 0.f};

  for (int nt = 0; nt < NT; ++nt) {
    v8f acc = {0.f, 0.f, 0.f, 0.f, 0.f, 0.f, 0.f, 0.f};
#pragma unroll
    for (int ks = 0; ks < KS; ++ks) {
      const _Float16* src = (KS == 16 && ks >= 8) ? ms : xs;
      const int kk = (ks & 7) * 32;
      union { v16h v; v8h p[2]; } a;
      a.p[0] = *(const v8h*)(src + m * DD + kk + half * 8);
      a.p[1] = *(const v8h*)(src + m * DD + kk + 16 + half * 8);
      v16h bf;
#pragma unroll
      for (int s = 0; s < 16; ++s)
        bf[s] = w1h[(size_t)(ks * 32 + (s < 8 ? half * 8 + s : 16 + half * 8 + (s - 8))) * N + nt * 16 + m];
      acc = __builtin_amdgcn_wmma_f32_16x16x32_f16(
          /*neg_a=*/false, a.v, /*neg_b=*/false, bf,
          /*c_mod=*/(short)0, acc, /*reuse_a=*/false, /*reuse_b=*/false);
    }
    const int n = nt * 16 + m;
    const float bias = b1[n];
    const float w2v = w2[n];
#pragma unroll
    for (int r = 0; r < 8; ++r) {
      float hv = acc[r] + bias;         // bias
      hv = hv > 0.f ? hv : 0.f;         // relu
      dot[r] += hv * w2v;               // layer-2 partial dot
    }
  }
#pragma unroll
  for (int r = 0; r < 8; ++r) {
    float v = dot[r];
#pragma unroll
    for (int off = 1; off < 16; off <<= 1) v += __shfl_xor(v, off, 16);
    if (m == 0) out[row0 + r + 8 * half] = 1.f / (1.f + __expf(-(v + b2)));
  }
}

// ---------------------------------------------------------------------------
// Kernel C: fused rot_w / trans_w / match_w. 128 threads = 4 waves,
// 16 rows per wave, 64 rows per block, 1024 blocks.
// ---------------------------------------------------------------------------
__global__ __launch_bounds__(128)
void mlp_fused_kernel(const float* __restrict__ qf, const float* __restrict__ tf,
                      const int* __restrict__ nn,
                      const _Float16* __restrict__ rw1h, const float* __restrict__ rb1,
                      const float* __restrict__ rw2, const float* __restrict__ rb2,
                      const _Float16* __restrict__ tw1h, const float* __restrict__ tb1,
                      const float* __restrict__ tw2, const float* __restrict__ tb2,
                      const _Float16* __restrict__ mn1h, const float* __restrict__ mb1,
                      const float* __restrict__ mw2, const float* __restrict__ mb2,
                      float* __restrict__ rot_out, float* __restrict__ trn_out,
                      float* __restrict__ mat_out) {
  __shared__ _Float16 xs[4][16][DD];
  __shared__ _Float16 ms[4][16][DD];
  const int w = threadIdx.x >> 5;
  const int lane = threadIdx.x & 31;
  const int row0 = blockIdx.x * 64 + w * 16;

  // Stage this wave's 16 activation rows (query + matched target) as f16.
  for (int rr = 0; rr < 16; ++rr) {
    const int row = row0 + rr;
    const int b = row >> 10;
    const int nv = nn[row];
    const float* xsrc = qf + (size_t)row * DD;
    const float* msrc = tf + (size_t)(b * TT + nv) * DD;
    for (int c = lane; c < DD; c += 32) {
      xs[w][rr][c] = (_Float16)xsrc[c];
      ms[w][rr][c] = (_Float16)msrc[c];
    }
  }
  __syncthreads();

  mlp_wave<8, 8>(&xs[w][0][0], nullptr, rw1h, rb1, rw2, rb2[0], rot_out, row0, lane);
  mlp_wave<8, 8>(&xs[w][0][0], nullptr, tw1h, tb1, tw2, tb2[0], trn_out, row0, lane);
  mlp_wave<16, 16>(&xs[w][0][0], &ms[w][0][0], mn1h, mb1, mw2, mb2[0], mat_out, row0, lane);
}

// ---------------------------------------------------------------------------
// SO(3) helpers (match reference semantics incl. small-angle branches)
// ---------------------------------------------------------------------------
__device__ __forceinline__ void so3_exp(const float w0, const float w1,
                                        const float w2, float R[9]) {
  const float th2 = w0 * w0 + w1 * w1 + w2 * w2;
  const float th = sqrtf(th2);
  if (th < 1e-5f) {
    // R = I + hat(w) + 0.5*hat(w)^2
    R[0] = 1.f - 0.5f * (w1 * w1 + w2 * w2);
    R[1] = -w2 + 0.5f * (w0 * w1);
    R[2] =  w1 + 0.5f * (w0 * w2);
    R[3] =  w2 + 0.5f * (w0 * w1);
    R[4] = 1.f - 0.5f * (w0 * w0 + w2 * w2);
    R[5] = -w0 + 0.5f * (w1 * w2);
    R[6] = -w1 + 0.5f * (w0 * w2);
    R[7] =  w0 + 0.5f * (w1 * w2);
    R[8] = 1.f - 0.5f * (w0 * w0 + w1 * w1);
  } else {
    const float thc = fmaxf(th, 1e-12f);
    const float kx = w0 / thc, ky = w1 / thc, kz = w2 / thc;
    const float st = sinf(thc), ct = cosf(thc), omc = 1.f - ct;
    R[0] = 1.f + omc * (-(ky * ky + kz * kz));
    R[1] = -st * kz + omc * (kx * ky);
    R[2] =  st * ky + omc * (kx * kz);
    R[3] =  st * kz + omc * (kx * ky);
    R[4] = 1.f + omc * (-(kx * kx + kz * kz));
    R[5] = -st * kx + omc * (ky * kz);
    R[6] = -st * ky + omc * (kx * kz);
    R[7] =  st * kx + omc * (ky * kz);
    R[8] = 1.f + omc * (-(kx * kx + ky * ky));
  }
}

// ---------------------------------------------------------------------------
// Kernel D: S=3 Riemannian refinement steps + output write
// ---------------------------------------------------------------------------
__global__ __launch_bounds__(256)
void refine_kernel(const float* __restrict__ R_pred, const float* __restrict__ t_pred,
                   const float* __restrict__ tgt_R, const float* __restrict__ tgt_t,
                   const int* __restrict__ nn, const float* __restrict__ ss,
                   const float* __restrict__ rot_w, const float* __restrict__ trn_w,
                   const float* __restrict__ mat_w, float* __restrict__ Rout,
                   float* __restrict__ tout) {
  const int row = blockIdx.x * 256 + threadIdx.x;
  if (row >= ROWS) return;
  const int b = row >> 10;
  const int nv = nn[row];

  float R[9], t[3], Rm[9], tm[3];
#pragma unroll
  for (int i = 0; i < 9; ++i) R[i] = R_pred[(size_t)row * 9 + i];
#pragma unroll
  for (int i = 0; i < 3; ++i) t[i] = t_pred[(size_t)row * 3 + i];
  const float* Rmp = tgt_R + (size_t)(b * TT + nv) * 9;
  const float* tmp = tgt_t + (size_t)(b * TT + nv) * 3;
#pragma unroll
  for (int i = 0; i < 9; ++i) Rm[i] = Rmp[i];
#pragma unroll
  for (int i = 0; i < 3; ++i) tm[i] = tmp[i];

  const float rw = rot_w[row];
  const float tw = trn_w[row];
  const float mw = mat_w[row];

  for (int s = 0; s < SS; ++s) {
    const float scale = ss[b * SS + s];
    // A = R^T @ Rm
    float A[9];
#pragma unroll
    for (int i = 0; i < 3; ++i)
#pragma unroll
      for (int j = 0; j < 3; ++j)
        A[i * 3 + j] = R[0 * 3 + i] * Rm[0 * 3 + j] +
                       R[1 * 3 + i] * Rm[1 * 3 + j] +
                       R[2 * 3 + i] * Rm[2 * 3 + j];
    // w = log_SO3(A)
    float tr = A[0] + A[4] + A[8];
    tr = fminf(fmaxf(tr, -1.f), 3.f);
    float c = fminf(fmaxf((tr - 1.f) * 0.5f, -1.f), 1.f);
    const float th = acosf(c);
    float v0 = 0.5f * (A[7] - A[5]);
    float v1 = 0.5f * (A[2] - A[6]);
    float v2 = 0.5f * (A[3] - A[1]);
    float w0, w1, w2;
    if (th < 1e-5f) {
      w0 = v0; w1 = v1; w2 = v2;
    } else {
      const float st = fmaxf(sinf(th), 1e-12f);
      const float f = th / st;
      w0 = v0 * f; w1 = v1 * f; w2 = v2 * f;
    }
    const float g = rw * mw * scale;
    float E[9];
    so3_exp(w0 * g, w1 * g, w2 * g, E);
    // R = R @ E
    float Rn[9];
#pragma unroll
    for (int i = 0; i < 3; ++i)
#pragma unroll
      for (int j = 0; j < 3; ++j)
        Rn[i * 3 + j] = R[i * 3 + 0] * E[0 * 3 + j] +
                        R[i * 3 + 1] * E[1 * 3 + j] +
                        R[i * 3 + 2] * E[2 * 3 + j];
#pragma unroll
    for (int i = 0; i < 9; ++i) R[i] = Rn[i];
    const float tg = tw * mw * scale;
#pragma unroll
    for (int i = 0; i < 3; ++i) t[i] += (tm[i] - t[i]) * tg;
  }

#pragma unroll
  for (int i = 0; i < 9; ++i) Rout[(size_t)row * 9 + i] = R[i];
#pragma unroll
  for (int i = 0; i < 3; ++i) tout[(size_t)row * 3 + i] = t[i];
}

// ---------------------------------------------------------------------------
// Host launcher
// ---------------------------------------------------------------------------
extern "C" void kernel_launch(void* const* d_in, const int* in_sizes, int n_in,
                              void* d_out, int out_size, void* d_ws, size_t ws_size,
                              hipStream_t stream) {
  (void)in_sizes; (void)n_in; (void)out_size; (void)ws_size;
  const float* R_pred  = (const float*)d_in[0];
  const float* t_pred  = (const float*)d_in[1];
  const float* p_boxes = (const float*)d_in[2];
  const float* qf      = (const float*)d_in[3];
  const float* tgt_R   = (const float*)d_in[4];
  const float* tgt_t   = (const float*)d_in[5];
  const float* t_boxes = (const float*)d_in[6];
  const float* tf      = (const float*)d_in[7];
  const float* sp_w1 = (const float*)d_in[8];
  const float* sp_b1 = (const float*)d_in[9];
  const float* sp_w2 = (const float*)d_in[10];
  const float* sp_b2 = (const float*)d_in[11];
  const float* rw_w1 = (const float*)d_in[12];
  const float* rw_b1 = (const float*)d_in[13];
  const float* rw_w2 = (const float*)d_in[14];
  const float* rw_b2 = (const float*)d_in[15];
  const float* tw_w1 = (const float*)d_in[16];
  const float* tw_b1 = (const float*)d_in[17];
  const float* tw_w2 = (const float*)d_in[18];
  const float* tw_b2 = (const float*)d_in[19];
  const float* mn_w1 = (const float*)d_in[20];
  const float* mn_b1 = (const float*)d_in[21];
  const float* mn_w2 = (const float*)d_in[22];
  const float* mn_b2 = (const float*)d_in[23];
  const float* gs    = (const float*)d_in[24];

  char* ws = (char*)d_ws;
  int*      nn    = (int*)(ws + OFF_NN);
  float*    ss    = (float*)(ws + OFF_SS);
  float*    rotw  = (float*)(ws + OFF_ROT);
  float*    trnw  = (float*)(ws + OFF_TRN);
  float*    matw  = (float*)(ws + OFF_MAT);
  _Float16* rw1h  = (_Float16*)(ws + OFF_RW1H);
  _Float16* tw1h  = (_Float16*)(ws + OFF_TW1H);
  _Float16* mn1h  = (_Float16*)(ws + OFF_MN1H);

  float* Rout = (float*)d_out;                       // (B,Q,3,3)
  float* tout = (float*)d_out + (size_t)ROWS * 9;    // (B,Q,3)

  // 1) f32 -> f16 weight conversion (layer-1 matrices)
  cvt_weights_kernel<<<(2 * DD * DD + 255) / 256, 256, 0, stream>>>(
      rw_w1, tw_w1, mn_w1, rw1h, tw1h, mn1h);
  // 2) step_scales (one block per batch)
  scales_kernel<<<BB, 256, 0, stream>>>(qf, sp_w1, sp_b1, sp_w2, sp_b2, gs, ss);
  // 3) nearest-neighbor match
  nn_kernel<<<ROWS / 256, 256, 0, stream>>>(p_boxes, t_boxes, nn);
  // 4) fused WMMA MLPs: rot_w / trans_w / match_w
  mlp_fused_kernel<<<ROWS / 64, 128, 0, stream>>>(
      qf, tf, nn, rw1h, rw_b1, rw_w2, rw_b2, tw1h, tw_b1, tw_w2, tw_b2,
      mn1h, mn_b1, mn_w2, mn_b2, rotw, trnw, matw);
  // 5) SO(3) refinement + output
  refine_kernel<<<ROWS / 256, 256, 0, stream>>>(
      R_pred, t_pred, tgt_R, tgt_t, nn, ss, rotw, trnw, matw, Rout, tout);
}
